// SDPAConv_31610959299273
// MI455X (gfx1250) — matrix operand order repaired and checked by reading, not compiled, not run on hardware
//
#include <hip/hip_runtime.h>

typedef __attribute__((ext_vector_type(16))) _Float16 v16h;
typedef __attribute__((ext_vector_type(8)))  float    v8f;

#define N_NODES 100000
#define CIN     256
#define COUT    256
#define KTAPS   9
#define MTILE   32
// B fragment store: [ (t*8+kc)*16 + ntile ][ lane*16 + v ]  (v16h per lane half-row)
#define BFRAG_ELEMS 512   // 32 lanes * 16 halves
#define BFRAG_TOTAL (KTAPS * 8 * 16 * BFRAG_ELEMS)  // 589824 halves

// ---------------------------------------------------------------------------
// Prep: convert fp32 weight [9,256,256] -> f16, pre-swizzled into WMMA
// B-fragment order so the main kernel loads 32 contiguous bytes per lane.
// B layout (16-bit, 32x16 KxN): lanes 0-15 -> N=lane, K=0..7 then 16..23;
//                               lanes 16-31 -> N=lane-16, K=8..15 then 24..31.
// ---------------------------------------------------------------------------
__global__ __launch_bounds__(256) void conv_weight_f16(
    const float* __restrict__ W, _Float16* __restrict__ Bf)
{
    int id = blockIdx.x * 256 + threadIdx.x;
    if (id >= BFRAG_TOTAL) return;
    int v    = id & 15;
    int lane = (id >> 4) & 31;
    int nt   = (id >> 9) & 15;
    int tkc  = id >> 13;            // 0..71 = t*8 + kc
    int t    = tkc >> 3;
    int kc   = tkc & 7;
    int klocal = (v < 8) ? ((lane >> 4) * 8 + v) : (16 + (lane >> 4) * 8 + (v - 8));
    int c = kc * 32 + klocal;       // GEMM K within tap  (C_in index)
    int d = nt * 16 + (lane & 15);  // output channel
    Bf[id] = (_Float16)W[((t * CIN) + c) * COUT + d];
}

// ---------------------------------------------------------------------------
// Main: out[32 nodes x 256 ch] per workgroup.
// 8 wave32: wave = rowBlock(2) x colStrip(4); each wave -> 16 rows x 64 cols.
// Double-buffered LDS A tile (32x256 f16) built per tap from gathered x.
// ---------------------------------------------------------------------------
__global__ __launch_bounds__(256, 2) void sdpa_conv_main(
    const float* __restrict__ x,
    const int*   __restrict__ nidx,
    const float* __restrict__ nw,
    const _Float16* __restrict__ Bf,
    const float* __restrict__ bias,
    float* __restrict__ out)
{
    __shared__ _Float16 Abuf[2][MTILE * CIN];   // 2 x 16 KB

    const int tid      = threadIdx.x;
    const int lane     = tid & 31;
    const int wave     = tid >> 5;
    const int nodeBase = blockIdx.x * MTILE;

    // --- staging role: 8 threads per row, 32 channels each ---
    const int r      = tid >> 3;   // 0..31 row of tile
    const int p      = tid & 7;    // 0..7  channel part
    const int node_r = nodeBase + r;

    // --- compute role ---
    const int rowBlock = wave >> 2;   // 0..1
    const int colStrip = wave & 3;    // 0..3
    const int m        = lane & 15;
    const int hlf      = lane >> 4;

    v8f acc[4] = {};
    float4 stage[8];

    auto do_load = [&](int t) -> float {
        int   srcNode;
        float w;
        if (t == 0) { srcNode = node_r; w = 1.0f; }
        else {
            srcNode = nidx[node_r * (KTAPS - 1) + (t - 1)];
            w       = nw  [node_r * (KTAPS - 1) + (t - 1)];
        }
        const float4* px = (const float4*)(x + (size_t)srcNode * CIN + p * 32);
#pragma unroll
        for (int i = 0; i < 8; ++i) stage[i] = px[i];   // 4x global_load_b128 pairs
        return w;
    };

    auto do_store = [&](int buf, float w) {
        _Float16* dst = &Abuf[buf][r * CIN + p * 32];
#pragma unroll
        for (int i = 0; i < 4; ++i) {
            struct alignas(16) { _Float16 h[8]; } pk;
            const float* f0 = (const float*)&stage[2 * i];
#pragma unroll
            for (int c = 0; c < 8; ++c) pk.h[c] = (_Float16)(w * f0[c]);
            *reinterpret_cast<float4*>(dst + i * 8) =
                *reinterpret_cast<const float4*>(&pk);   // ds_store_b128
        }
    };

    auto do_compute = [&](int buf, int t) {
        const _Float16* Abase = &Abuf[buf][0];
#pragma unroll
        for (int kc = 0; kc < 8; ++kc) {
            // A layout (16-bit 16x32): lanes 0-15: K0..7 then K16..23;
            //                          lanes 16-31: K8..15 then K24..31.
            const _Float16* ap =
                Abase + (rowBlock * 16 + m) * CIN + kc * 32 + hlf * 8;
            v16h a;
            *reinterpret_cast<float4*>(&a)       = *reinterpret_cast<const float4*>(ap);
            *(reinterpret_cast<float4*>(&a) + 1) = *reinterpret_cast<const float4*>(ap + 16);
#pragma unroll
            for (int i = 0; i < 4; ++i) {
                const int ntg = colStrip * 4 + i;
                const _Float16* bp =
                    Bf + (size_t)(((t * 8 + kc) * 16 + ntg) << 9) + lane * 16;
                v16h b;
                *reinterpret_cast<float4*>(&b)       = *reinterpret_cast<const float4*>(bp);
                *(reinterpret_cast<float4*>(&b) + 1) = *reinterpret_cast<const float4*>(bp + 8);
                acc[i] = __builtin_amdgcn_wmma_f32_16x16x32_f16(
                    false, a, false, b, (short)0, acc[i], false, false);
            }
        }
    };

    // --- software-pipelined tap loop (double-buffered A tile) ---
    float w0 = do_load(0);
    do_store(0, w0);
    __syncthreads();

#pragma unroll 1
    for (int t = 0; t < KTAPS; ++t) {
        float wnext = 0.0f;
        if (t + 1 < KTAPS) wnext = do_load(t + 1);   // prefetch next tap's gather
        do_compute(t & 1, t);
        if (t + 1 < KTAPS) do_store((t + 1) & 1, wnext);
        __syncthreads();
    }

    // --- epilogue: bias add + store (C/D layout: VGPR j -> row hlf*8+j) ---
#pragma unroll
    for (int i = 0; i < 4; ++i) {
        const int col = colStrip * 64 + i * 16 + m;
        const float bv = bias[col];
#pragma unroll
        for (int j = 0; j < 8; ++j) {
            const int node = nodeBase + rowBlock * 16 + hlf * 8 + j;
            out[(size_t)node * COUT + col] = acc[i][j] + bv;
        }
    }
}

extern "C" void kernel_launch(void* const* d_in, const int* in_sizes, int n_in,
                              void* d_out, int out_size, void* d_ws, size_t ws_size,
                              hipStream_t stream)
{
    const float* x    = (const float*)d_in[0];
    const int*   nidx = (const int*)  d_in[1];
    const float* nw   = (const float*)d_in[2];
    const float* W    = (const float*)d_in[3];
    const float* bias = (const float*)d_in[4];
    float*       out  = (float*)d_out;
    _Float16*    Bf   = (_Float16*)d_ws;     // 1.15 MB of scratch

    conv_weight_f16<<<(BFRAG_TOTAL + 255) / 256, 256, 0, stream>>>(W, Bf);
    sdpa_conv_main<<<N_NODES / MTILE, 256, 0, stream>>>(x, nidx, nw, Bf, bias, out);
}
